// MSDTHGTEncoder_34024730919095
// MI455X (gfx1250) — compile-verified, matrix operand order checked by reading, not run on hardware
//
#include <hip/hip_runtime.h>
#include <hip/hip_bf16.h>
#include <math.h>

// ---------------------------------------------------------------------------
// MSDTHGT encoder for MI455X (gfx1250, wave32). All dense contractions run on
// v_wmma_f32_16x16x32_f16 (f16 multiplicands, f32 accumulate). One wave per
// 16x16 output tile. Fragment loads are branch-free: each lane's 16 elements
// are two contiguous 8-float runs -> four b128 loads; out-of-range handled by
// row/K clamping (clamped garbage only reaches never-stored rows or is
// multiplied by zeroed softmax pad columns).
// ---------------------------------------------------------------------------

typedef __attribute__((ext_vector_type(16))) _Float16 v16h;
typedef __attribute__((ext_vector_type(8)))  float    v8f;

#define DEV __device__ __forceinline__

DEV v8f wmma16(v16h a, v16h b, v8f c) {
  // D = A(16x32,f16) x B(32x16,f16) + C(16x16,f32)
  return __builtin_amdgcn_wmma_f32_16x16x32_f16(false, a, false, b, (short)0, c,
                                                false, false);
}

// A-operand (or B^T) 16x32 fragment from row-major f32, branch-free.
// Lane layout (ISA 7.12.2): row = lane&15, K base = k0 + (lane>>4)*8;
// elements = k[kb..kb+7] then k[kb+16..kb+23] (two contiguous float4 pairs).
// rclamp = nrows-1: rows beyond are clamped (duplicated), caller guarantees
// such rows never reach a stored output.
DEV v16h frag_a(const float* src, long ld, int row0, int k0, int rclamp) {
  int l = threadIdx.x & 31;
  int r = row0 + (l & 15);
  r = r < rclamp ? r : rclamp;
  const float* p = src + (long)r * ld + k0 + ((l >> 4) << 3);
  float4 a = *(const float4*)(p);
  float4 b = *(const float4*)(p + 4);
  float4 c = *(const float4*)(p + 16);
  float4 d = *(const float4*)(p + 20);
  v16h f;
  f[0] = (_Float16)a.x;  f[1] = (_Float16)a.y;
  f[2] = (_Float16)a.z;  f[3] = (_Float16)a.w;
  f[4] = (_Float16)b.x;  f[5] = (_Float16)b.y;
  f[6] = (_Float16)b.z;  f[7] = (_Float16)b.w;
  f[8] = (_Float16)c.x;  f[9] = (_Float16)c.y;
  f[10] = (_Float16)c.z; f[11] = (_Float16)c.w;
  f[12] = (_Float16)d.x; f[13] = (_Float16)d.y;
  f[14] = (_Float16)d.z; f[15] = (_Float16)d.w;
  return f;
}

// B-operand fragment from a K x N row-major matrix (element (n,k)<-src[k*ld+n]).
// kclamp = nk-1; clamped K rows must be multiplied by zero A columns.
DEV v16h frag_bT(const float* src, long ld, int col0, int k0, int kclamp) {
  int l = threadIdx.x & 31;
  int n = col0 + (l & 15);
  int kb = k0 + ((l >> 4) << 3);
  v16h f;
#pragma unroll
  for (int v = 0; v < 8; ++v) {
    int kk = kb + ((v & 3) << 1) + ((v >> 2) << 4);
    int kA = kk < kclamp ? kk : kclamp;
    int kB = (kk + 1) < kclamp ? (kk + 1) : kclamp;
    f[2 * v] = (_Float16)src[(long)kA * ld + n];
    f[2 * v + 1] = (_Float16)src[(long)kB * ld + n];
  }
  return f;
}

// Store a C/D tile where only rows < 24 are valid (T=24 attention tiles).
// Validity depends only on the 8-row group -> one uniform per-lane branch.
DEV void store_tile_rows24(float* base, long rstride, int row0, int col0, v8f c) {
  int l = threadIdx.x & 31;
  int n = col0 + (l & 15);
  int rb = (l >> 4) << 3;
  if (row0 + rb < 24) {
#pragma unroll
    for (int v = 0; v < 8; ++v)
      base[(long)(row0 + rb + v) * rstride + n] = c[v];
  }
}

// ---------------------------------------------------------------------------
// Generic GEMM: C(MxN) = act(A(MxK) @ W^T(NxK,row-major) + bias [+ resid])
// Requires M%16==0, N%16==0, K%32==0 (all call sites satisfy this).
// One wave per 16x16 tile, 4 waves / block.
// ---------------------------------------------------------------------------
__global__ void gemm_bias_kernel(const float* __restrict__ A, long lda,
                                 const float* __restrict__ W, long ldw,
                                 const float* __restrict__ bias,
                                 const float* __restrict__ resid, long ldr,
                                 float* __restrict__ C, long ldc,
                                 int M, int Nc, int K, int relu) {
  int wave = blockIdx.x * (blockDim.x >> 5) + (threadIdx.x >> 5);
  int tilesN = Nc >> 4;
  int tilesM = M >> 4;
  if (wave >= tilesM * tilesN) return;
  int tm = wave / tilesN, tn = wave % tilesN;
  int row0 = tm << 4, col0 = tn << 4;
  v8f acc = {};
  for (int kk = 0; kk < K; kk += 32) {
    if (kk + 32 < K) {
      __builtin_prefetch(A + (long)row0 * lda + kk + 32, 0, 3);
      __builtin_prefetch(W + (long)col0 * ldw + kk + 32, 0, 3);
    }
    v16h a = frag_a(A, lda, row0, kk, M - 1);
    v16h b = frag_a(W, ldw, col0, kk, Nc - 1);
    acc = wmma16(a, b, acc);
  }
  int l = threadIdx.x & 31;
  int n = col0 + (l & 15);
  int m0 = row0 + ((l >> 4) << 3);
  float bv = bias ? bias[n] : 0.0f;
#pragma unroll
  for (int v = 0; v < 8; ++v) {
    float x = acc[v] + bv;
    if (resid) x += resid[(long)(m0 + v) * ldr + n];
    if (relu) x = fmaxf(x, 0.0f);
    C[(long)(m0 + v) * ldc + n] = x;
  }
}

static void launch_gemm(const float* A, long lda, const float* W, long ldw,
                        const float* bias, const float* resid, long ldr,
                        float* C, long ldc, int M, int Nc, int K, int relu,
                        hipStream_t st) {
  int tiles = (M / 16) * (Nc / 16);
  gemm_bias_kernel<<<(tiles + 3) / 4, 128, 0, st>>>(A, lda, W, ldw, bias, resid,
                                                    ldr, C, ldc, M, Nc, K, relu);
}

// ---------------------------------------------------------------------------
// GTU causal conv + gate. Writes directly in the reference's "ro" flat layout:
// ro_flat[((n*2 + b)*128 + o)*24 + t]  (i.e. (N,B,d,T) viewed as (T,B*N,d)).
// ---------------------------------------------------------------------------
__global__ void gtu_conv_kernel(const float* __restrict__ X, long sB, long sN,
                                int cin, int kk, const float* __restrict__ W,
                                const float* __restrict__ bias,
                                float* __restrict__ RO) {
  long i = (long)blockIdx.x * blockDim.x + threadIdx.x;
  if (i >= 2L * 200 * 24 * 128) return;
  int o = (int)(i & 127);
  long r = i >> 7;
  int t = (int)(r % 24); r /= 24;
  int n = (int)(r % 200);
  int b = (int)(r / 200);
  const float* xb = X + b * sB + n * sN;
  float ct = bias[o], cs = bias[o + 128];
  for (int j = 0; j < kk; ++j) {
    int tt = t + j - (kk - 1);
    if (tt < 0) continue;
    const float4* xr = (const float4*)(xb + (long)tt * cin);
    const float* w0 = W + ((long)o * cin) * kk + j;
    const float* w1 = W + ((long)(o + 128) * cin) * kk + j;
    int q = cin >> 2;
    for (int c4 = 0; c4 < q; ++c4) {
      float4 xv = xr[c4];
      int cbase = c4 << 2;
      ct += w0[(long)(cbase + 0) * kk] * xv.x + w0[(long)(cbase + 1) * kk] * xv.y +
            w0[(long)(cbase + 2) * kk] * xv.z + w0[(long)(cbase + 3) * kk] * xv.w;
      cs += w1[(long)(cbase + 0) * kk] * xv.x + w1[(long)(cbase + 1) * kk] * xv.y +
            w1[(long)(cbase + 2) * kk] * xv.z + w1[(long)(cbase + 3) * kk] * xv.w;
    }
  }
  float g = tanhf(ct) * (1.0f / (1.0f + __expf(-cs)));
  RO[((long)(n * 2 + b) * 128 + o) * 24 + t] = g;
}

// ---------------------------------------------------------------------------
// Single-head sequence attention (T=24, d=128) on packed QKV rows (t*S+s, 384).
// One wave per batch element s; scores + softmax + PV fused through 4KB LDS.
// ---------------------------------------------------------------------------
__global__ void attn_seq_kernel(const float* __restrict__ QKV, int S,
                                float* __restrict__ O) {
  __shared__ __align__(16) float P[32 * 32];
  int s = blockIdx.x;
  const float* base = QKV + (long)s * 384;
  long ld = (long)S * 384;
  v8f s00 = {}, s01 = {}, s10 = {}, s11 = {};
  for (int kk = 0; kk < 128; kk += 32) {
    v16h q0 = frag_a(base, ld, 0, kk, 23);
    v16h q1 = frag_a(base, ld, 16, kk, 23);
    v16h k0 = frag_a(base + 128, ld, 0, kk, 23);
    v16h k1 = frag_a(base + 128, ld, 16, kk, 23);
    s00 = wmma16(q0, k0, s00);
    s01 = wmma16(q0, k1, s01);
    s10 = wmma16(q1, k0, s10);
    s11 = wmma16(q1, k1, s11);
  }
  {  // zero P (pad region must be exactly 0 for the PV K-padding)
    float4* P4 = (float4*)P;
    for (int i = threadIdx.x; i < 256; i += 32) P4[i] = float4{0.f, 0.f, 0.f, 0.f};
  }
  __syncthreads();
  const float sc = 0.08838834764831845f;  // 1/sqrt(128)
  int l = threadIdx.x & 31;
  {  // masked score store: rows via uniform 8-row-group guard, cols via select
    int rb = (l >> 4) << 3;
    int nlo = l & 15, nhi = 16 + (l & 15);
    float slo = (nlo < 24) ? sc : 0.0f;   // (always true for nlo, kept for symmetry)
    float shi = (nhi < 24) ? sc : 0.0f;
#pragma unroll
    for (int v = 0; v < 8; ++v) {       // tiles with row0 = 0: rows rb+v < 24 always
      P[(rb + v) * 32 + nlo] = s00[v] * slo;
      P[(rb + v) * 32 + nhi] = s01[v] * shi;
    }
    if (rb < 8) {                        // tiles with row0 = 16: valid iff 16+rb < 24
#pragma unroll
      for (int v = 0; v < 8; ++v) {
        P[(16 + rb + v) * 32 + nlo] = s10[v] * slo;
        P[(16 + rb + v) * 32 + nhi] = s11[v] * shi;
      }
    }
  }
  __syncthreads();
  if (l < 24) {
    float mx = -1e30f;
    for (int u = 0; u < 24; ++u) mx = fmaxf(mx, P[l * 32 + u]);
    float sum = 0.0f;
    for (int u = 0; u < 24; ++u) {
      float e = __expf(P[l * 32 + u] - mx);
      P[l * 32 + u] = e;
      sum += e;
    }
    float inv = 1.0f / sum;
    for (int u = 0; u < 24; ++u) P[l * 32 + u] *= inv;
  }
  __syncthreads();
  v16h a0 = frag_a(P, 32, 0, 0, 31);
  v16h a1 = frag_a(P, 32, 16, 0, 31);
  const float* Vb = base + 256;
  float* Ob = O + (long)s * 128;
  long ldo = (long)S * 128;
  v8f z = {};
  for (int nt = 0; nt < 8; ++nt) {
    v16h bv = frag_bT(Vb, ld, nt * 16, 0, 23);  // K rows >=24 clamped; P pad cols are 0
    v8f c0 = wmma16(a0, bv, z);
    v8f c1 = wmma16(a1, bv, z);
    store_tile_rows24(Ob, ldo, 0, nt * 16, c0);
    store_tile_rows24(Ob, ldo, 16, nt * 16, c1);
  }
}

// ---------------------------------------------------------------------------
// Residual + LayerNorm over d=128 (one wave per row).
// ---------------------------------------------------------------------------
__global__ void ln_residual_kernel(const float* __restrict__ X,
                                   const float* __restrict__ R,
                                   const float* __restrict__ g,
                                   const float* __restrict__ bt,
                                   float* __restrict__ O, int rows) {
  int r = blockIdx.x * (blockDim.x >> 5) + (threadIdx.x >> 5);
  if (r >= rows) return;
  int l = threadIdx.x & 31;
  const float* x = X + (long)r * 128;
  const float* rr = R + (long)r * 128;
  float v0[4], s = 0.0f, s2 = 0.0f;
#pragma unroll
  for (int i = 0; i < 4; ++i) {
    float t = x[l + 32 * i] + rr[l + 32 * i];
    v0[i] = t; s += t; s2 += t * t;
  }
  for (int o = 16; o; o >>= 1) { s += __shfl_xor(s, o); s2 += __shfl_xor(s2, o); }
  float mean = s * (1.0f / 128.0f);
  float var = s2 * (1.0f / 128.0f) - mean * mean;
  float inv = rsqrtf(var + 1e-5f);
#pragma unroll
  for (int i = 0; i < 4; ++i) {
    int c = l + 32 * i;
    O[(long)r * 128 + c] = (v0[i] - mean) * inv * g[c] + bt[c];
  }
}

// Gather X3_i (ro layout) -> SBUF rows (t*1200 + (i*200+n)*2 + b).
__global__ void sbuf_gather_kernel(const float* __restrict__ X30,
                                   const float* __restrict__ X31,
                                   const float* __restrict__ X32,
                                   float* __restrict__ SBUF) {
  long i = (long)blockIdx.x * blockDim.x + threadIdx.x;
  if (i >= 28800L * 128) return;
  int c = (int)(i & 127);
  long r = i >> 7;
  int s3 = (int)(r % 1200);
  int t = (int)(r / 1200);
  int b = s3 & 1;
  int np = s3 >> 1;
  int scx = np / 200, n = np % 200;
  const float* X3 = (scx == 0) ? X30 : ((scx == 1) ? X31 : X32);
  SBUF[i] = X3[((long)t * 400 + b * 200 + n) * 128 + c];
}

// Mean over the 3 scales -> y[b,n,t,c] with caller-provided strides.
__global__ void ymean_kernel(const float* __restrict__ AOUT, float* __restrict__ Y,
                             long sBo, long sNo) {
  long i = (long)blockIdx.x * blockDim.x + threadIdx.x;
  if (i >= 2L * 200 * 24 * 128) return;
  int c = (int)(i & 127);
  long r = i >> 7;
  int t = (int)(r % 24); r /= 24;
  int n = (int)(r % 200);
  int b = (int)(r / 200);
  float acc = 0.0f;
  for (int s = 0; s < 3; ++s)
    acc += AOUT[((long)t * 1200 + (s * 200 + n) * 2 + b) * 128 + c];
  Y[b * sBo + n * sNo + (long)t * 128 + c] = acc * (1.0f / 3.0f);
}

// tb[e] = sum_d E[e,d]*w[d], with E row 0 zeroed (padding_idx).
__global__ void type_bias_kernel(const float* __restrict__ E,
                                 const float* __restrict__ w,
                                 float* __restrict__ tb) {
  int e = threadIdx.x >> 5, l = threadIdx.x & 31;
  if (e >= 3) return;
  float s = 0.0f;
  for (int c = l; c < 128; c += 32) s += E[e * 128 + c] * w[c];
  for (int o = 16; o; o >>= 1) s += __shfl_xor(s, o);
  if (l == 0) tb[e] = (e == 0) ? 0.0f : s;
}

__global__ void edge_bias_kernel(const int* __restrict__ ept,
                                 const int* __restrict__ ept_r,
                                 const float* __restrict__ tb,
                                 const float* __restrict__ ebb,
                                 float* __restrict__ EB) {
  long i = (long)blockIdx.x * blockDim.x + threadIdx.x;
  if (i >= 320000L) return;
  int dir = i >= 160000L;
  long e = i - (dir ? 160000L : 0L);
  const int* ep = (dir ? ept_r : ept) + e * 4;
  float s = 0.25f * (tb[ep[0]] + tb[ep[1]] + tb[ep[2]] + tb[ep[3]]);
  EB[i] = s + ebb[0];
}

// ---------------------------------------------------------------------------
// CHGAN per-node-type QKV: one wave per (n,h); (24x32)@(32x32) x3 on WMMA.
// ---------------------------------------------------------------------------
__global__ void chgan_qkv_kernel(const float* __restrict__ NF,
                                 const int* __restrict__ node_type,
                                 const float* __restrict__ WQ,
                                 const float* __restrict__ WK,
                                 const float* __restrict__ WV,
                                 float* __restrict__ Q, float* __restrict__ K,
                                 float* __restrict__ V) {
  int id = blockIdx.x;
  int h = id & 3, n = id >> 2;
  int ty = node_type[n];
  const float* A = NF + ((long)n * 24) * 128 + h * 32;
  v16h a0 = frag_a(A, 128, 0, 0, 23);
  v16h a1 = frag_a(A, 128, 16, 0, 23);
  const float* Ws[3] = {WQ, WK, WV};
  float* Os[3] = {Q, K, V};
  v8f z = {};
#pragma unroll
  for (int p = 0; p < 3; ++p) {
    const float* W = Ws[p] + (long)(ty * 4 + h) * 1024;  // (32x32) d x c
    v16h b0 = frag_bT(W, 32, 0, 0, 31);
    v16h b1 = frag_bT(W, 32, 16, 0, 31);
    v8f c00 = wmma16(a0, b0, z), c01 = wmma16(a0, b1, z);
    v8f c10 = wmma16(a1, b0, z), c11 = wmma16(a1, b1, z);
    float* ob = Os[p] + (long)h * 24 * 400 * 32 + (long)n * 32;  // row t, stride 400*32
    store_tile_rows24(ob, 400 * 32, 0, 0, c00);
    store_tile_rows24(ob, 400 * 32, 0, 16, c01);
    store_tile_rows24(ob, 400 * 32, 16, 0, c10);
    store_tile_rows24(ob, 400 * 32, 16, 16, c11);
  }
}

// ---------------------------------------------------------------------------
// CHGAN fused graph attention: wave per (dir,h,t, 16-row node tile).
// Scores (16x400) biased/scaled/masked in LDS, softmax, then P@V on WMMA
// with K padded 400->416 (zero probs).
// ---------------------------------------------------------------------------
__global__ void chgan_attn_kernel(const float* __restrict__ Q,
                                  const float* __restrict__ K,
                                  const float* __restrict__ V,
                                  const float* __restrict__ EB,
                                  const float* __restrict__ eplF,
                                  const int* __restrict__ mskF,
                                  const float* __restrict__ eplR,
                                  const int* __restrict__ mskR,
                                  float* __restrict__ OF) {
  __shared__ __align__(16) float P[16 * 416];
  int bid = blockIdx.x;
  int nt = bid % 25; bid /= 25;
  int t = bid % 24; bid /= 24;
  int h = bid & 3;
  int dir = bid >> 2;
  int n0 = nt * 16;
  long hoff = ((long)h * 24 + t) * 400 * 32;
  const float* Qh = Q + hoff;
  const float* Kh = K + hoff;
  const float* Vh = V + hoff;
  const float* EBd = EB + (long)dir * 160000;
  const float* EPL = dir ? eplR : eplF;
  const int* MSK = dir ? mskR : mskF;
  int l = threadIdx.x & 31;
  v16h aq = frag_a(Qh, 32, n0, 0, 399);
  const float scl = 0.1767766952966369f;  // 1/sqrt(32)
  v8f z = {};
  for (int mt = 0; mt < 25; ++mt) {
    v16h bk = frag_a(Kh, 32, mt * 16, 0, 399);
    v8f acc = wmma16(aq, bk, z);
    int m = mt * 16 + (l & 15);
    int rb = (l >> 4) << 3;
#pragma unroll
    for (int v = 0; v < 8; ++v) {
      int n = n0 + rb + v;
      long e = (long)n * 400 + m;
      float sv = acc[v] * scl + EBd[e];
      sv *= __expf(0.5f * (EPL[e] - 1.0f));
      if (MSK[e] == 0) sv = -1e30f;
      P[(rb + v) * 416 + m] = sv;
    }
  }
  __syncthreads();
  {  // softmax over 400 cols; 2 lanes per row
    int r = l >> 1, half = l & 1;
    int m0 = half * 200;
    float mx = -1e30f;
    for (int m = m0; m < m0 + 200; ++m) mx = fmaxf(mx, P[r * 416 + m]);
    mx = fmaxf(mx, __shfl_xor(mx, 1));
    float sum = 0.0f;
    for (int m = m0; m < m0 + 200; ++m) {
      float e = __expf(P[r * 416 + m] - mx);
      P[r * 416 + m] = e;
      sum += e;
    }
    sum += __shfl_xor(sum, 1);
    float inv = 1.0f / sum;
    for (int m = m0; m < m0 + 200; ++m) P[r * 416 + m] *= inv;
    if (half == 0)
      for (int m = 400; m < 416; ++m) P[r * 416 + m] = 0.0f;
  }
  __syncthreads();
  v8f acc0 = {}, acc1 = {};
  for (int ks = 0; ks < 416; ks += 32) {
    v16h ap = frag_a(P, 416, 0, ks, 15);
    v16h b0 = frag_bT(Vh, 32, 0, ks, 399);   // K rows >=400 clamped; P pad cols are 0
    v16h b1 = frag_bT(Vh, 32, 16, ks, 399);
    acc0 = wmma16(ap, b0, acc0);
    acc1 = wmma16(ap, b1, acc1);
  }
  // OF[n, t, dir*128 + h*32 + c], rows (n,t) with feature stride 256
  float* ob = OF + (long)t * 256 + dir * 128 + h * 32;
  long rs = 24L * 256;  // stride in n
  int c = l & 15;
  int rb = (l >> 4) << 3;
#pragma unroll
  for (int v = 0; v < 8; ++v) {
    int n = n0 + rb + v;
    ob[(long)n * rs + c] = acc0[v];
    ob[(long)n * rs + 16 + c] = acc1[v];
  }
}

// ---------------------------------------------------------------------------
// Final LayerNorm over (Nk=200, D=128) per (b,t), then ReLU, scatter to out.
// ---------------------------------------------------------------------------
__global__ void final_ln_relu_kernel(const float* __restrict__ Z,
                                     const float* __restrict__ g,
                                     const float* __restrict__ bt,
                                     float* __restrict__ OUT, int node_off) {
  __shared__ float sh[8], sh2[8], stats[2];
  int b = blockIdx.x / 24, t = blockIdx.x % 24;
  float s = 0.0f, s2 = 0.0f;
  for (int i = threadIdx.x; i < 25600; i += blockDim.x) {
    int nk = i >> 7, c = i & 127;
    float v = Z[(((long)b * 200 + nk) * 24 + t) * 128 + c];
    s += v; s2 += v * v;
  }
  for (int o = 16; o; o >>= 1) { s += __shfl_xor(s, o); s2 += __shfl_xor(s2, o); }
  if ((threadIdx.x & 31) == 0) { sh[threadIdx.x >> 5] = s; sh2[threadIdx.x >> 5] = s2; }
  __syncthreads();
  if (threadIdx.x == 0) {
    float ts = 0.0f, ts2 = 0.0f;
    for (int w = 0; w < 8; ++w) { ts += sh[w]; ts2 += sh2[w]; }
    float mean = ts * (1.0f / 25600.0f);
    float var = ts2 * (1.0f / 25600.0f) - mean * mean;
    stats[0] = mean;
    stats[1] = rsqrtf(var + 1e-5f);
  }
  __syncthreads();
  float mean = stats[0], inv = stats[1];
  for (int i = threadIdx.x; i < 25600; i += blockDim.x) {
    int nk = i >> 7, c = i & 127;
    float v = Z[(((long)b * 200 + nk) * 24 + t) * 128 + c];
    float o = (v - mean) * inv * g[i] + bt[i];
    OUT[(((long)b * 400 + node_off + nk) * 24 + t) * 128 + c] = fmaxf(o, 0.0f);
  }
}

// ---------------------------------------------------------------------------
// Host-side orchestration
// ---------------------------------------------------------------------------
struct TLp {
  const float *ff1_b, *ff1_w, *ff2_b, *ff2_w, *in_b, *in_w;
  const float *ln1_b, *ln1_g, *ln2_b, *ln2_g, *out_b, *out_w;
};
struct MSWTp {
  const float* gtu_b[3];
  const float* gtu_w[3];
  const float *mha_in_b, *mha_in_w, *mha_out_b, *mha_out_w;
  TLp tl[3];
};

static inline const float* F(void* const* d, int i) { return (const float*)d[i]; }

static MSWTp make_mswt(void* const* d, int base) {
  MSWTp m;
  int i = base;
  for (int s = 0; s < 3; ++s) { m.gtu_b[s] = F(d, i++); m.gtu_w[s] = F(d, i++); }
  m.mha_in_b = F(d, i++); m.mha_in_w = F(d, i++);
  m.mha_out_b = F(d, i++); m.mha_out_w = F(d, i++);
  for (int s = 0; s < 3; ++s) {
    TLp& t = m.tl[s];
    t.ff1_b = F(d, i++); t.ff1_w = F(d, i++);
    t.ff2_b = F(d, i++); t.ff2_w = F(d, i++);
    t.in_b = F(d, i++);  t.in_w = F(d, i++);
    t.ln1_b = F(d, i++); t.ln1_g = F(d, i++);
    t.ln2_b = F(d, i++); t.ln2_g = F(d, i++);
    t.out_b = F(d, i++); t.out_w = F(d, i++);
  }
  return m;
}

// workspace layout (floats)
static const long X3SZ = 9600L * 128;             // 1,228,800
static const long NF_ELEMS = 2L * 400 * 24 * 128; // 2,457,600
static const long OFF_RO = 3 * X3SZ;
static const long OFF_BUF1 = OFF_RO + X3SZ;
static const long OFF_BUF2 = OFF_BUF1 + 9600L * 512;
static const long OFF_BUF3 = OFF_BUF2 + X3SZ;
static const long OFF_SBUF = 3 * X3SZ;            // phase B aliases phase A
static const long OFF_QKV3 = OFF_SBUF + 28800L * 128;
static const long OFF_ATTO3 = OFF_QKV3 + 28800L * 384;
static const long OFF_AOUT = OFF_ATTO3 + 28800L * 128;
static const long OFF_ZBUF = OFF_AOUT + 28800L * 128;

static void run_mswt(const MSWTp& P, const float* X, long sB, long sN, int cin,
                     float* Y, long sBo, long sNo, float* pool, hipStream_t st) {
  float* X3[3] = {pool, pool + X3SZ, pool + 2 * X3SZ};
  float* RO = pool + OFF_RO;
  float* B1 = pool + OFF_BUF1;
  float* B2 = pool + OFF_BUF2;
  float* B3 = pool + OFF_BUF3;
  for (int i = 0; i < 3; ++i) {
    gtu_conv_kernel<<<4800, 256, 0, st>>>(X, sB, sN, cin, i + 1, P.gtu_w[i], P.gtu_b[i], RO);
    const TLp& t = P.tl[i];
    launch_gemm(RO, 128, t.in_w, 128, t.in_b, nullptr, 0, B1, 384, 9600, 384, 128, 0, st);
    attn_seq_kernel<<<400, 32, 0, st>>>(B1, 400, B2);
    launch_gemm(B2, 128, t.out_w, 128, t.out_b, nullptr, 0, B1, 128, 9600, 128, 128, 0, st);
    ln_residual_kernel<<<2400, 128, 0, st>>>(RO, B1, t.ln1_g, t.ln1_b, B3, 9600);
    launch_gemm(B3, 128, t.ff1_w, 128, t.ff1_b, nullptr, 0, B1, 512, 9600, 512, 128, 1, st);
    launch_gemm(B1, 512, t.ff2_w, 512, t.ff2_b, nullptr, 0, B2, 128, 9600, 128, 512, 0, st);
    ln_residual_kernel<<<2400, 128, 0, st>>>(B3, B2, t.ln2_g, t.ln2_b, X3[i], 9600);
  }
  float* SBUF = pool + OFF_SBUF;
  float* QKV3 = pool + OFF_QKV3;
  float* AT3 = pool + OFF_ATTO3;
  float* AO = pool + OFF_AOUT;
  sbuf_gather_kernel<<<14400, 256, 0, st>>>(X3[0], X3[1], X3[2], SBUF);
  launch_gemm(SBUF, 128, P.mha_in_w, 128, P.mha_in_b, nullptr, 0, QKV3, 384, 28800, 384, 128, 0, st);
  attn_seq_kernel<<<1200, 32, 0, st>>>(QKV3, 1200, AT3);
  launch_gemm(AT3, 128, P.mha_out_w, 128, P.mha_out_b, nullptr, 0, AO, 128, 28800, 128, 128, 0, st);
  ymean_kernel<<<4800, 256, 0, st>>>(AO, Y, sBo, sNo);
}

extern "C" void kernel_launch(void* const* d_in, const int* in_sizes, int n_in,
                              void* d_out, int out_size, void* d_ws, size_t ws_size,
                              hipStream_t stream) {
  (void)in_sizes; (void)n_in; (void)out_size; (void)ws_size;
  const float* x0 = F(d_in, 0);
  const float* x1 = F(d_in, 1);
  const int* node_type = (const int*)d_in[3];
  const int* ept = (const int*)d_in[4];
  const float* epl = F(d_in, 5);
  const int* msk = (const int*)d_in[6];
  const int* ept_r = (const int*)d_in[7];
  const float* epl_r = F(d_in, 8);
  const int* msk_r = (const int*)d_in[9];
  // params: jax tree_flatten (alphabetical dict keys)
  const float* cE = F(d_in, 10);
  const float* cK = F(d_in, 11);
  const float* cQ = F(d_in, 12);
  const float* cV = F(d_in, 13);
  const float* c_eb_b = F(d_in, 14);
  const float* c_eb_w = F(d_in, 15);
  const float* c_out_b = F(d_in, 16);
  const float* c_out_w = F(d_in, 17);
  const float* ln1_b = F(d_in, 18);
  const float* ln1_g = F(d_in, 19);
  const float* ln2_b = F(d_in, 20);
  const float* ln2_g = F(d_in, 21);
  MSWTp m1 = make_mswt(d_in, 22);
  MSWTp m2 = make_mswt(d_in, 68);

  float* ws = (float*)d_ws;
  float* XNF = ws;                  // (B,400,24,128) concat of mswt1 outputs
  float* CH = XNF + NF_ELEMS;       // chgan output
  float* POOL = CH + NF_ELEMS;

  // ---- stage 1: y0/y1 = mswt1(x0/x1) written into XNF at node offsets ----
  run_mswt(m1, x0, 200L * 24 * 32, 24L * 32, 32, XNF, 400L * 24 * 128, 24L * 128, POOL, stream);
  run_mswt(m1, x1, 200L * 24 * 32, 24L * 32, 32, XNF + 200L * 24 * 128,
           400L * 24 * 128, 24L * 128, POOL, stream);

  // ---- stage 2: chgan per batch ----
  float* TB = POOL;
  float* EB = POOL + 32;
  float* Qc = EB + 320000;
  float* Kc = Qc + 1228800;
  float* Vc = Kc + 1228800;
  float* OFb = Vc + 1228800;  // (400,24,256) concat fwd|rev
  type_bias_kernel<<<1, 128, 0, stream>>>(cE, c_eb_w, TB);
  edge_bias_kernel<<<(320000 + 255) / 256, 256, 0, stream>>>(ept, ept_r, TB, c_eb_b, EB);
  for (int b = 0; b < 2; ++b) {
    const float* nf = XNF + (long)b * 1228800;
    chgan_qkv_kernel<<<1600, 32, 0, stream>>>(nf, node_type, cQ, cK, cV, Qc, Kc, Vc);
    chgan_attn_kernel<<<4800, 32, 0, stream>>>(Qc, Kc, Vc, EB, epl, msk, epl_r, msk_r, OFb);
    launch_gemm(OFb, 256, c_out_w, 256, c_out_b, nf, 128, CH + (long)b * 1228800,
                128, 9600, 128, 256, 0, stream);
  }

  // ---- stage 3: mswt2 on node slices + final grouped LN + ReLU ----
  float* ZBUF = POOL + OFF_ZBUF;
  for (int sl = 0; sl < 2; ++sl) {
    run_mswt(m2, CH + (long)sl * 200 * 24 * 128, 400L * 24 * 128, 24L * 128, 128,
             ZBUF, 200L * 24 * 128, 24L * 128, POOL, stream);
    final_ln_relu_kernel<<<48, 256, 0, stream>>>(ZBUF, sl ? ln2_g : ln1_g,
                                                 sl ? ln2_b : ln1_b,
                                                 (float*)d_out, sl * 200);
  }
}
